// QuantizedLinearLayer_90598040142545
// MI455X (gfx1250) — compile-verified, matrix operand order checked by reading, not run on hardware
//
#include <hip/hip_runtime.h>
#include <stdint.h>

typedef __attribute__((ext_vector_type(16))) _Float16 v16h;
typedef __attribute__((ext_vector_type(8)))  _Float16 v8h;
typedef __attribute__((ext_vector_type(4)))  _Float16 v4h;
typedef __attribute__((ext_vector_type(8)))  float    v8f;
typedef __attribute__((ext_vector_type(4)))  float    v4f;
typedef __attribute__((ext_vector_type(4)))  int      v4i;

namespace {
constexpr int kBatch = 4, kSeq = 2048, kIN = 4096, kOUT = 4096;
constexpr int M = kBatch * kSeq;  // 8192
constexpr int N = kOUT;           // 4096
constexpr int K = kIN;            // 4096
constexpr int BM = 128, BN = 128, BK = 32;
constexpr int NT = K / BK;        // 128 K-steps
// LDS row stride: 40 halfs = 80 bytes. Multiple of 16B (b128 alignment kept),
// = 20 banks, so 16 consecutive rows map to 16 distinct banks -> conflict-free.
constexpr int LDT = BK + 8;
}

union Frag { v16h v; v8h h[2]; };

// ---------------------------------------------------------------------------
// Kernel 1: W_f16[o][i] = mask ? outlier : (float)q * scale[o]   (f32 -> f16)
// ---------------------------------------------------------------------------
__global__ __launch_bounds__(256) void reconstruct_w_f16(
    const int* __restrict__ q, const float* __restrict__ scales,
    const float* __restrict__ outw, const unsigned char* __restrict__ mask,
    _Float16* __restrict__ w16)
{
  const int i0 = (blockIdx.x * 256 + threadIdx.x) * 8;
  if (i0 >= N * K) return;
  v4i q0 = *(const v4i*)(q + i0);
  v4i q1 = *(const v4i*)(q + i0 + 4);
  v4f o0 = *(const v4f*)(outw + i0);
  v4f o1 = *(const v4f*)(outw + i0 + 4);
  uint64_t m8 = *(const uint64_t*)(mask + i0);   // 8 bool bytes, 8B-aligned
  const float s = scales[i0 >> 12];              // IN = 4096, 8 | 4096
  v8h r;
#pragma unroll
  for (int j = 0; j < 8; ++j) {
    int   qv = (j < 4) ? q0[j] : q1[j - 4];
    float ov = (j < 4) ? o0[j] : o1[j - 4];
    bool  mm = ((m8 >> (8 * j)) & 0xffull) != 0;
    float f  = mm ? ov : (float)qv * s;
    r[j] = (_Float16)f;
  }
  *(v8h*)(w16 + i0) = r;
}

// ---------------------------------------------------------------------------
// Async/LDS helpers
// ---------------------------------------------------------------------------
__device__ __forceinline__ uint32_t lds_off(const void* p) {
  // Low 32 bits of a shared-aperture flat pointer == addrspace(3) byte offset.
  return (uint32_t)(uintptr_t)p;
}

__device__ __forceinline__ void wait_async0() {
  asm volatile("s_wait_asynccnt 0x0" ::: "memory");
}

// W tile 128x32 f16: 2 x b128 per thread, global -> LDS via the CDNA5 async
// copy engine (ASYNCcnt). No VGPR data staging, overlaps the WMMA block.
__device__ __forceinline__ void async_w_tile(const _Float16* __restrict__ W,
                                             int n0, int kbase, int tid,
                                             _Float16* __restrict__ lw)
{
#pragma unroll
  for (int p = 0; p < 2; ++p) {
    int lin = (p * 256 + tid) * 8;
    int row = lin >> 5, col = lin & 31;
    uint64_t ga = (uint64_t)(uintptr_t)(W + (size_t)(n0 + row) * K + kbase + col);
    uint32_t la = lds_off(lw + row * LDT + col);
    asm volatile("global_load_async_to_lds_b128 %0, %1, off"
                 :: "v"(la), "v"(ga) : "memory");
  }
}

// X tile 128x32 f32: global -> VGPR (needs f32->f16 convert on the way to LDS)
__device__ __forceinline__ void fetch_x(const float* __restrict__ X,
                                        int m0, int kbase, int tid, v4f xr[4])
{
#pragma unroll
  for (int p = 0; p < 4; ++p) {
    int lin = (p * 256 + tid) * 4;
    int row = lin >> 5, col = lin & 31;
    xr[p] = *(const v4f*)(X + (size_t)(m0 + row) * K + kbase + col);
  }
}

__device__ __forceinline__ void stage_x(_Float16* __restrict__ lx,
                                        int tid, const v4f xr[4])
{
#pragma unroll
  for (int p = 0; p < 4; ++p) {              // convert f32->f16, ds_store_b64
    int lin = (p * 256 + tid) * 4;
    int row = lin >> 5, col = lin & 31;
    v4h h;
#pragma unroll
    for (int j = 0; j < 4; ++j) h[j] = (_Float16)xr[p][j];
    *(v4h*)(lx + row * LDT + col) = h;
  }
}

// ---------------------------------------------------------------------------
// Kernel 2: Y[M][N] = X[M][K] * W[N][K]^T + bias, f16 WMMA, f32 accumulate.
// Block tile 128x128, BK=32; 8 wave32 arranged 4(M) x 2(N); each wave owns
// 2x4 16x16 accumulator tiles. Double-buffered LDS; X via reg staging,
// W via global_load_async_to_lds_b128; one barrier per K-step.
// ---------------------------------------------------------------------------
__global__ __launch_bounds__(256) void wmma_gemm_f16(
    const float* __restrict__ X, const _Float16* __restrict__ W,
    const float* __restrict__ bias, float* __restrict__ Y)
{
  __shared__ _Float16 ldsX[2][BM * LDT];   // 2 x 10240 B
  __shared__ _Float16 ldsW[2][BN * LDT];   // 2 x 10240 B  -> 40 KB total

  const int tid  = threadIdx.x;
  const int lane = tid & 31;
  const int wave = tid >> 5;
  const int wm   = wave & 3;    // wave M offset: wm*32  (2 tiles)
  const int wn   = wave >> 2;   // wave N offset: wn*64  (4 tiles)
  const int m0   = blockIdx.y * BM;
  const int n0   = blockIdx.x * BN;

  // Fragment addressing (identical for A and B: W is [N][K] row-major, which
  // matches the ISA B-matrix per-lane layout directly -> no transpose).
  const int frow = lane & 15;          // matrix row (A) / W row = B column (B)
  const int koff = (lane >> 4) * 8;    // K split: lanes 0-15 -> 0..7/16..23,
                                       //          lanes 16-31 -> 8..15/24..31
  v8f acc[2][4];
#pragma unroll
  for (int i = 0; i < 2; ++i)
#pragma unroll
    for (int j = 0; j < 4; ++j) acc[i][j] = (v8f)(0.0f);

  v4f xr[4];

  // Prologue: stage tile 0
  fetch_x(X, m0, 0, tid, xr);
  async_w_tile(W, n0, 0, tid, ldsW[0]);
  stage_x(ldsX[0], tid, xr);
  wait_async0();            // drain own asyncs before signaling the barrier
  __syncthreads();

  int cur = 0;
  for (int kt = 0; kt < NT; ++kt) {
    if (kt + 1 < NT) {
      fetch_x(X, m0, (kt + 1) * BK, tid, xr);
      async_w_tile(W, n0, (kt + 1) * BK, tid, ldsW[cur ^ 1]);
    }

    Frag a[2], b[4];
#pragma unroll
    for (int i = 0; i < 2; ++i) {
      const _Float16* base = &ldsX[cur][(wm * 32 + i * 16 + frow) * LDT];
      a[i].h[0] = *(const v8h*)(base + koff);        // ds_load_b128
      a[i].h[1] = *(const v8h*)(base + 16 + koff);   // ds_load_b128
    }
#pragma unroll
    for (int j = 0; j < 4; ++j) {
      const _Float16* base = &ldsW[cur][(wn * 64 + j * 16 + frow) * LDT];
      b[j].h[0] = *(const v8h*)(base + koff);
      b[j].h[1] = *(const v8h*)(base + 16 + koff);
    }

#pragma unroll
    for (int i = 0; i < 2; ++i)
#pragma unroll
      for (int j = 0; j < 4; ++j)
        acc[i][j] = __builtin_amdgcn_wmma_f32_16x16x32_f16(
            /*neg_a=*/false, a[i].v, /*neg_b=*/false, b[j].v,
            /*c_mod=*/(short)0, acc[i][j],
            /*reuse_a=*/false, /*reuse_b=*/false);

    if (kt + 1 < NT) {
      stage_x(ldsX[cur ^ 1], tid, xr);
      wait_async0();        // own async W tile placed before barrier signal
      __syncthreads();
      cur ^= 1;
    }
  }

  // Epilogue: C/D layout -> VGPR r holds M = r + 8*(lane>=16), N = lane&15.
  const int mrow = m0 + wm * 32 + (lane >> 4) * 8;
  const int ncol = n0 + wn * 64 + (lane & 15);
#pragma unroll
  for (int i = 0; i < 2; ++i) {
    const int mbase = mrow + i * 16;
#pragma unroll
    for (int j = 0; j < 4; ++j) {
      const int nn = ncol + j * 16;
      const float bv = bias[nn];
#pragma unroll
      for (int r = 0; r < 8; ++r)
        Y[(size_t)(mbase + r) * N + nn] = acc[i][j][r] + bv;
    }
  }
}

// ---------------------------------------------------------------------------
// Inputs (setup_inputs order): input f32, q_weights i32, scales f32,
// outlier_weights f32, outlier_mask bool(1B), bias f32. Output f32 [B,S,OUT].
// d_ws: 32 MB for reconstructed f16 weights.
// ---------------------------------------------------------------------------
extern "C" void kernel_launch(void* const* d_in, const int* in_sizes, int n_in,
                              void* d_out, int out_size, void* d_ws, size_t ws_size,
                              hipStream_t stream) {
  const float*         x      = (const float*)d_in[0];
  const int*           q      = (const int*)d_in[1];
  const float*         scales = (const float*)d_in[2];
  const float*         outw   = (const float*)d_in[3];
  const unsigned char* mask   = (const unsigned char*)d_in[4];
  const float*         bias   = (const float*)d_in[5];
  float*               y      = (float*)d_out;
  _Float16*            w16    = (_Float16*)d_ws;   // needs OUT*IN*2 = 32 MB

  (void)in_sizes; (void)n_in; (void)out_size; (void)ws_size;

  const int reconBlocks = (N * K) / (8 * 256);     // 8192
  reconstruct_w_f16<<<reconBlocks, 256, 0, stream>>>(q, scales, outw, mask, w16);

  dim3 grid(N / BN, M / BM);                       // (32, 64)
  wmma_gemm_f16<<<grid, 256, 0, stream>>>(x, w16, bias, y);
}
